// SelfAttention_88124138979516
// MI455X (gfx1250) — matrix operand order repaired and checked by reading, not compiled, not run on hardware
//
#include <hip/hip_runtime.h>

// ---------------------------------------------------------------------------
// Self-attention (B=2, T=2048, K=512, H=8) for gfx1250 using bf16 WMMA.
// Pipeline: cvt(x) + transpose-cvt(W*) -> QKV GEMMs -> flash attention -> out GEMM
// Round 3: fix async-to-LDS builtin pointer types (v4i* in AS1/AS3).
// ---------------------------------------------------------------------------

typedef __attribute__((ext_vector_type(16))) __bf16 v16bf;
typedef __attribute__((ext_vector_type(8)))  float  v8f;
typedef unsigned int v4u __attribute__((ext_vector_type(4)));
typedef int          v8i __attribute__((ext_vector_type(8)));
typedef int          v4i __attribute__((ext_vector_type(4)));

#define GLOBAL_AS __attribute__((address_space(1)))
#define LOCAL_AS  __attribute__((address_space(3)))

#if __has_builtin(__builtin_amdgcn_global_load_async_to_lds_b128) && \
    __has_builtin(__builtin_amdgcn_s_wait_asynccnt)
#define HAS_ASYNC_LDS 1
#else
#define HAS_ASYNC_LDS 0
#endif

#if __has_builtin(__builtin_amdgcn_tensor_load_to_lds) && \
    __has_builtin(__builtin_amdgcn_s_wait_tensorcnt)
#define HAS_TDM 1
#else
#define HAS_TDM 0
#endif

#define WMMA_BF16(a, b, c) \
  __builtin_amdgcn_wmma_f32_16x16x32_bf16(false, (a), false, (b), (short)0, (c), false, false)

static __device__ __forceinline__ v8f zero_v8f() {
  v8f z;
#pragma unroll
  for (int i = 0; i < 8; ++i) z[i] = 0.0f;
  return z;
}

union Frag16 { v16bf v; uint4 u[2]; };

// A fragment: 16x32 bf16 tile, source row-major (M rows, K contiguous).
// Lane L<16 holds row M=L, K {0..7, 16..23}; lane L>=16 holds row M=L-16, K {8..15, 24..31}.
static __device__ __forceinline__ v16bf load_a_frag(const __bf16* p, int stride, int lane) {
  const int row = lane & 15;
  const int kb  = (lane >> 4) << 3;          // 0 or 8
  Frag16 f;
  f.u[0] = *(const uint4*)(p + (size_t)row * stride + kb);
  f.u[1] = *(const uint4*)(p + (size_t)row * stride + kb + 16);
  return f.v;
}

// B fragment: 32x16 bf16 tile from an N-major (pre-transposed) source:
// row n of source = column n of B; lane L<16 holds K {0..7, 16..23} of column L, lanes>=16 K {8..15, 24..31}.
static __device__ __forceinline__ v16bf load_b_frag(const __bf16* p, int stride, int lane) {
  const int n  = lane & 15;
  const int kb = (lane >> 4) << 3;           // 0 or 8
  Frag16 f;
  f.u[0] = *(const uint4*)(p + (size_t)n * stride + kb);
  f.u[1] = *(const uint4*)(p + (size_t)n * stride + kb + 16);
  return f.v;
}

// ---------------------------------------------------------------------------
// fp32 -> bf16 elementwise (8 elements / thread)
// ---------------------------------------------------------------------------
__global__ __launch_bounds__(256)
void cvt_bf16_kernel(const float* __restrict__ in, __bf16* __restrict__ out, int n8) {
  int i = blockIdx.x * blockDim.x + threadIdx.x;
  if (i >= n8) return;
  const float4* p = (const float4*)in + (size_t)i * 2;
  float4 x0 = p[0], x1 = p[1];
  __bf16 o[8] = {(__bf16)x0.x, (__bf16)x0.y, (__bf16)x0.z, (__bf16)x0.w,
                 (__bf16)x1.x, (__bf16)x1.y, (__bf16)x1.z, (__bf16)x1.w};
  *(uint4*)(out + (size_t)i * 8) = *(const uint4*)o;
}

// ---------------------------------------------------------------------------
// fp32 [R,C] row-major -> bf16 [C,R] (N-major weights). Block (32,8), 32x32 tiles.
// ---------------------------------------------------------------------------
__global__ __launch_bounds__(256)
void transpose_cvt_kernel(const float* __restrict__ W, __bf16* __restrict__ Wt, int R, int C) {
  __shared__ float tile[32][33];
  const int tx = threadIdx.x, ty = threadIdx.y;
  const int c0 = blockIdx.x * 32, r0 = blockIdx.y * 32;
#pragma unroll
  for (int k = 0; k < 4; ++k)
    tile[ty + 8 * k][tx] = W[(size_t)(r0 + ty + 8 * k) * C + (c0 + tx)];
  __syncthreads();
#pragma unroll
  for (int k = 0; k < 4; ++k)
    Wt[(size_t)(c0 + ty + 8 * k) * R + (r0 + tx)] = (__bf16)tile[tx][ty + 8 * k];
}

// ---------------------------------------------------------------------------
// GEMM  C = A[M,Kd] * Bt[N,Kd]^T  (both bf16, K contiguous), epilogue writes
// bf16 into head-major [B=2, H=8, T=2048, 512] layout (for Q/K/V).
// ---------------------------------------------------------------------------
__global__ __launch_bounds__(256)
void gemm_qkv_kernel(const __bf16* __restrict__ A, const __bf16* __restrict__ Bt,
                     __bf16* __restrict__ dst, int Kd) {
  const int lane = threadIdx.x & 31;
  const int wave = threadIdx.x >> 5;
  const int m0 = blockIdx.x * 128 + (wave >> 2) * 64;
  const int n0 = blockIdx.y * 128 + (wave & 3) * 32;

  v8f acc[4][2];
#pragma unroll
  for (int i = 0; i < 4; ++i)
#pragma unroll
    for (int j = 0; j < 2; ++j) acc[i][j] = zero_v8f();

  for (int k0 = 0; k0 < Kd; k0 += 32) {
    v16bf a[4], b[2];
#pragma unroll
    for (int i = 0; i < 4; ++i)
      a[i] = load_a_frag(A + (size_t)(m0 + 16 * i) * Kd + k0, Kd, lane);
#pragma unroll
    for (int j = 0; j < 2; ++j)
      b[j] = load_b_frag(Bt + (size_t)(n0 + 16 * j) * Kd + k0, Kd, lane);
#pragma unroll
    for (int i = 0; i < 4; ++i)
#pragma unroll
      for (int j = 0; j < 2; ++j)
        acc[i][j] = WMMA_BF16(a[i], b[j], acc[i][j]);
  }

  const int n = lane & 15, hi = lane >> 4;
#pragma unroll
  for (int i = 0; i < 4; ++i) {
#pragma unroll
    for (int j = 0; j < 2; ++j) {
      const int cg = n0 + 16 * j + n;            // 0..4095 = h*512 + d
      const int h_ = cg >> 9, d_ = cg & 511;
#pragma unroll
      for (int r = 0; r < 8; ++r) {
        const int mg = m0 + 16 * i + r + 8 * hi; // 0..4095 = b*2048 + t
        const int b_ = mg >> 11, t_ = mg & 2047;
        dst[((size_t)((b_ << 3) + h_) * 2048 + t_) * 512 + d_] = (__bf16)acc[i][j][r];
      }
    }
  }
}

// ---------------------------------------------------------------------------
// GEMM with bias, f32 output row-major (final projection).
// ---------------------------------------------------------------------------
__global__ __launch_bounds__(256)
void gemm_out_kernel(const __bf16* __restrict__ A, const __bf16* __restrict__ Bt,
                     const float* __restrict__ bias, float* __restrict__ out,
                     int Kd, int N) {
  const int lane = threadIdx.x & 31;
  const int wave = threadIdx.x >> 5;
  const int m0 = blockIdx.x * 128 + (wave >> 2) * 64;
  const int n0 = blockIdx.y * 128 + (wave & 3) * 32;

  v8f acc[4][2];
#pragma unroll
  for (int i = 0; i < 4; ++i)
#pragma unroll
    for (int j = 0; j < 2; ++j) acc[i][j] = zero_v8f();

  for (int k0 = 0; k0 < Kd; k0 += 32) {
    v16bf a[4], b[2];
#pragma unroll
    for (int i = 0; i < 4; ++i)
      a[i] = load_a_frag(A + (size_t)(m0 + 16 * i) * Kd + k0, Kd, lane);
#pragma unroll
    for (int j = 0; j < 2; ++j)
      b[j] = load_b_frag(Bt + (size_t)(n0 + 16 * j) * Kd + k0, Kd, lane);
#pragma unroll
    for (int i = 0; i < 4; ++i)
#pragma unroll
      for (int j = 0; j < 2; ++j)
        acc[i][j] = WMMA_BF16(a[i], b[j], acc[i][j]);
  }

  const int n = lane & 15, hi = lane >> 4;
#pragma unroll
  for (int i = 0; i < 4; ++i) {
#pragma unroll
    for (int j = 0; j < 2; ++j) {
      const int cg = n0 + 16 * j + n;
      const float bv = bias[cg];
#pragma unroll
      for (int r = 0; r < 8; ++r) {
        const int mg = m0 + 16 * i + r + 8 * hi;
        out[(size_t)mg * N + cg] = acc[i][j][r] + bv;
      }
    }
  }
}

// ---------------------------------------------------------------------------
// TDM helper: 2D tile load Global->LDS, data_size=8B units.
//   widthU  = tile/tensor row length in 8-byte units
//   rows    = tile/tensor rows
// Descriptor per CDNA5 ISA 8.3/8.4 (group0: count/lds/global/type, group1:
// data_size, tensor dims, tile dims, dim0 stride).
// ---------------------------------------------------------------------------
#if HAS_TDM
static __device__ __forceinline__ void tdm_load_2d(const void* gsrc, const void* lds,
                                                   unsigned widthU, unsigned rows) {
  const unsigned long long ga = (unsigned long long)gsrc;
  const unsigned la = (unsigned)(unsigned long long)(LOCAL_AS const char*)lds;
  v4u g0;
  g0[0] = 1u;                                   // count=1, user descriptor
  g0[1] = la;                                   // lds_addr (bytes)
  g0[2] = (unsigned)ga;                         // global_addr[31:0]
  g0[3] = (unsigned)(ga >> 32) | (2u << 30);    // global_addr[56:32] | type=2
  v8i g1;
  g1[0] = 3 << 16;                              // data_size = 3 (8 bytes)
  g1[1] = (int)(widthU << 16);                  // tensor_dim0[15:0]
  g1[2] = (int)(rows << 16);                    // tensor_dim0[31:16]=0 | tensor_dim1[15:0]
  g1[3] = (int)(widthU << 16);                  // tensor_dim1[31:16]=0 | tile_dim0
  g1[4] = (int)rows;                            // tile_dim1 | tile_dim2=0
  g1[5] = (int)widthU;                          // tensor_dim0_stride[31:0]
  g1[6] = 0;                                    // stride[47:32] | tensor_dim1_stride[15:0]
  g1[7] = 0;
  v4i g2, g3;
#pragma unroll
  for (int i = 0; i < 4; ++i) { g2[i] = 0; g3[i] = 0; }
#if defined(__clang_major__) && (__clang_major__ >= 23)
  v8i g4;
#pragma unroll
  for (int i = 0; i < 8; ++i) g4[i] = 0;
  __builtin_amdgcn_tensor_load_to_lds(g0, g1, g2, g3, g4, 0);
#else
  __builtin_amdgcn_tensor_load_to_lds(g0, g1, g2, g3, 0);
#endif
}
#endif

// ---------------------------------------------------------------------------
// Flash attention, one (b,h) per blockIdx.y, 128 query rows per blockIdx.x.
// 8 waves x 16 query rows each, full d=512 accumulated in registers.
// LDS: Q panel 128KB + K tile 32KB + V^T tile 32KB + P staging 8KB = 200KB
// (exploits the 320KB/WGP LDS of CDNA5; one workgroup per WGP).
// ---------------------------------------------------------------------------
#define LOG2E_F 1.44269504088896f

__global__ __launch_bounds__(256)
void attn_fa_kernel(const __bf16* __restrict__ Qh, const __bf16* __restrict__ Kh,
                    const __bf16* __restrict__ Vh, __bf16* __restrict__ Aout) {
  extern __shared__ char smem[];
  __bf16* Qs  = (__bf16*)smem;            // [128][512]
  __bf16* Ks  = Qs  + 128 * 512;          // [32][512]
  __bf16* Vts = Ks  + 32 * 512;           // [512][32]  (transposed V tile)
  __bf16* Ps  = Vts + 512 * 32;           // [8][16][32] per-wave P staging

  const int tid  = threadIdx.x;
  const int wave = tid >> 5;
  const int lane = tid & 31;
  const int bh   = blockIdx.y;            // b*8 + h
  const int q0   = blockIdx.x * 128;

  const __bf16* Qg = Qh + ((size_t)bh * 2048 + q0) * 512;
  const __bf16* Kg = Kh + (size_t)bh * 2048 * 512;
  const __bf16* Vg = Vh + (size_t)bh * 2048 * 512;

  // --- stage the 128x512 Q panel once: TDM if available, else manual copy ---
#if HAS_TDM
  if (wave == 0) {
    tdm_load_2d(Qg, Qs, /*widthU=*/128, /*rows=*/128);   // 128 rows x 1024 B
    __builtin_amdgcn_s_wait_tensorcnt(0);
  }
#else
  {
    const uint4* src = (const uint4*)Qg;
    uint4* dst = (uint4*)Qs;
#pragma unroll 4
    for (int t = 0; t < 64; ++t) dst[tid + 256 * t] = src[tid + 256 * t];
  }
#endif

  v8f acc[32];
#pragma unroll
  for (int d = 0; d < 32; ++d) acc[d] = zero_v8f();
  float mrun[8], lrun[8];
#pragma unroll
  for (int r = 0; r < 8; ++r) { mrun[r] = -3.0e38f; lrun[r] = 0.0f; }

  const float scale = 0.044194173824159216f;   // 512^-0.5 (= (k^-0.25)^2)
  const int n  = lane & 15;
  const int hi = lane >> 4;
  __bf16* pw = Ps + wave * (16 * 32);

  for (int j = 0; j < 2048; j += 32) {
    // --- cooperative K tile (async->LDS when available) + transposed V tile ---
    {
      const uint4* ks = (const uint4*)(Kg + (size_t)j * 512);
      uint4* kd = (uint4*)Ks;
#if HAS_ASYNC_LDS
#pragma unroll
      for (int t = 0; t < 8; ++t) {
        const int u = tid + 256 * t;
        __builtin_amdgcn_global_load_async_to_lds_b128(
            (GLOBAL_AS v4i*)(ks + u), (LOCAL_AS v4i*)(kd + u), 0, 0);
      }
#else
#pragma unroll
      for (int t = 0; t < 8; ++t) kd[tid + 256 * t] = ks[tid + 256 * t];
#endif
      const uint4* vs = (const uint4*)(Vg + (size_t)j * 512);
#pragma unroll
      for (int t = 0; t < 8; ++t) {
        const int u  = tid + 256 * t;       // over 2048 uint4 = 32x512 bf16
        uint4 val    = vs[u];
        const int tv = u >> 6;              // kv row 0..31
        const int dv = (u & 63) << 3;       // d col base
        const __bf16* e = (const __bf16*)&val;
#pragma unroll
        for (int q = 0; q < 8; ++q) Vts[(size_t)(dv + q) * 32 + tv] = e[q];
      }
      // prefetch next iteration's K and V tiles into cache while we compute
      if (j + 32 < 2048) {
        const char* nk = (const char*)(Kg + (size_t)(j + 32) * 512);
        const char* nv = (const char*)(Vg + (size_t)(j + 32) * 512);
        __builtin_prefetch(nk + tid * 128, 0, 1);
        __builtin_prefetch(nv + tid * 128, 0, 1);
      }
#if HAS_ASYNC_LDS
      __builtin_amdgcn_s_wait_asynccnt(0);
#endif
    }
    __syncthreads();

    // --- S = Q * K^T for this wave's 16 rows (two 16x16 tiles: kv 0..15, 16..31)
    v8f s0 = zero_v8f(), s1 = zero_v8f();
    const __bf16* qrow = Qs + (size_t)(wave * 16) * 512;
#pragma unroll 4
    for (int kk = 0; kk < 512; kk += 32) {
      v16bf aq  = load_a_frag(qrow + kk, 512, lane);
      v16bf bk0 = load_b_frag(Ks + kk, 512, lane);
      v16bf bk1 = load_b_frag(Ks + (size_t)16 * 512 + kk, 512, lane);
      s0 = WMMA_BF16(aq, bk0, s0);
      s1 = WMMA_BF16(aq, bk1, s1);
    }

    // --- online softmax (row m = r + 8*hi lives in one 16-lane half) ---
    float p0[8], p1[8], alpha[8];
#pragma unroll
    for (int r = 0; r < 8; ++r) {
      float v0 = s0[r] * scale, v1 = s1[r] * scale;
      float mx = fmaxf(v0, v1);
      mx = fmaxf(mx, __shfl_xor(mx, 1));
      mx = fmaxf(mx, __shfl_xor(mx, 2));
      mx = fmaxf(mx, __shfl_xor(mx, 4));
      mx = fmaxf(mx, __shfl_xor(mx, 8));
      const float mnew = fmaxf(mrun[r], mx);
      const float a = exp2f((mrun[r] - mnew) * LOG2E_F);
      p0[r] = exp2f((v0 - mnew) * LOG2E_F);
      p1[r] = exp2f((v1 - mnew) * LOG2E_F);
      float rs = p0[r] + p1[r];
      rs += __shfl_xor(rs, 1);
      rs += __shfl_xor(rs, 2);
      rs += __shfl_xor(rs, 4);
      rs += __shfl_xor(rs, 8);
      lrun[r] = lrun[r] * a + rs;
      mrun[r] = mnew;
      alpha[r] = a;
    }
#pragma unroll
    for (int d = 0; d < 32; ++d)
#pragma unroll
      for (int r = 0; r < 8; ++r) acc[d][r] *= alpha[r];

    // --- P (C/D layout) -> LDS -> A-fragment layout ---
#pragma unroll
    for (int r = 0; r < 8; ++r) {
      pw[(r + 8 * hi) * 32 + n]      = (__bf16)p0[r];
      pw[(r + 8 * hi) * 32 + n + 16] = (__bf16)p1[r];
    }
    v16bf ap = load_a_frag(pw, 32, lane);

    // --- O += P * V over all 32 d-subtiles ---
#pragma unroll
    for (int d = 0; d < 32; ++d) {
      v16bf bv = load_b_frag(Vts + (size_t)(d * 16) * 32, 32, lane);
      acc[d] = WMMA_BF16(ap, bv, acc[d]);
    }
    __syncthreads();
  }

  // --- normalize and write [b, t, h*512 + d] bf16 ---
  float inv[8];
#pragma unroll
  for (int r = 0; r < 8; ++r) inv[r] = 1.0f / lrun[r];
  const int b_ = bh >> 3, h_ = bh & 7;
#pragma unroll
  for (int d = 0; d < 32; ++d) {
    const int dd = d * 16 + n;
#pragma unroll
    for (int r = 0; r < 8; ++r) {
      const int t_ = q0 + wave * 16 + r + 8 * hi;
      Aout[((size_t)(b_ * 2048 + t_)) * 4096 + h_ * 512 + dd] = (__bf16)(acc[d][r] * inv[r]);
    }
  }
}

// ---------------------------------------------------------------------------
// Host launch
// ---------------------------------------------------------------------------
extern "C" void kernel_launch(void* const* d_in, const int* in_sizes, int n_in,
                              void* d_out, int out_size, void* d_ws, size_t ws_size,
                              hipStream_t stream) {
  (void)in_sizes; (void)n_in; (void)out_size; (void)ws_size;
  const float* x  = (const float*)d_in[0];   // [2,2048,512]
  const float* Wq = (const float*)d_in[1];   // [512,4096]
  const float* Wk = (const float*)d_in[2];
  const float* Wv = (const float*)d_in[3];
  const float* Wu = (const float*)d_in[4];   // [4096,512]
  const float* bu = (const float*)d_in[5];   // [512]
  float* out = (float*)d_out;                // [2,2048,512] f32

  char* ws = (char*)d_ws;
  __bf16* xb  = (__bf16*)ws;  ws += (size_t)4096 * 512 * 2;
  __bf16* Wqt = (__bf16*)ws;  ws += (size_t)4096 * 512 * 2;   // [4096][512]
  __bf16* Wkt = (__bf16*)ws;  ws += (size_t)4096 * 512 * 2;
  __bf16* Wvt = (__bf16*)ws;  ws += (size_t)4096 * 512 * 2;
  __bf16* Wut = (__bf16*)ws;  ws += (size_t)512 * 4096 * 2;   // [512][4096]
  __bf16* Qh  = (__bf16*)ws;  ws += (size_t)16 * 2048 * 512 * 2;
  __bf16* Kh  = (__bf16*)ws;  ws += (size_t)16 * 2048 * 512 * 2;
  __bf16* Vh  = (__bf16*)ws;  ws += (size_t)16 * 2048 * 512 * 2;
  __bf16* Ao  = (__bf16*)ws;  ws += (size_t)4096 * 4096 * 2;

  // 1) conversions
  cvt_bf16_kernel<<<1024, 256, 0, stream>>>(x, xb, 4096 * 512 / 8);
  {
    dim3 g(4096 / 32, 512 / 32), b(32, 8);
    transpose_cvt_kernel<<<g, b, 0, stream>>>(Wq, Wqt, 512, 4096);
    transpose_cvt_kernel<<<g, b, 0, stream>>>(Wk, Wkt, 512, 4096);
    transpose_cvt_kernel<<<g, b, 0, stream>>>(Wv, Wvt, 512, 4096);
  }
  {
    dim3 g(512 / 32, 4096 / 32), b(32, 8);
    transpose_cvt_kernel<<<g, b, 0, stream>>>(Wu, Wut, 4096, 512);
  }

  // 2) QKV projections: [4096,512] x [512,4096] -> head-major bf16
  {
    dim3 g(4096 / 128, 4096 / 128);
    gemm_qkv_kernel<<<g, 256, 0, stream>>>(xb, Wqt, Qh, 512);
    gemm_qkv_kernel<<<g, 256, 0, stream>>>(xb, Wkt, Kh, 512);
    gemm_qkv_kernel<<<g, 256, 0, stream>>>(xb, Wvt, Vh, 512);
  }

  // 3) flash attention
  {
    dim3 g(2048 / 128, 16);
    size_t lds = (size_t)(128 * 512 + 32 * 512 + 512 * 32 + 8 * 16 * 32) * 2;  // 204800 B
    attn_fa_kernel<<<g, 256, lds, stream>>>(Qh, Kh, Vh, Ao);
  }

  // 4) output projection: [4096,4096] x [4096,512] + bias -> f32
  {
    dim3 g(4096 / 128, 512 / 128);
    gemm_out_kernel<<<g, 256, 0, stream>>>(Ao, Wut, bu, out, 4096, 512);
  }
}
